// RelationGraphConv_77129022701791
// MI455X (gfx1250) — matrix operand order repaired and checked by reading, not compiled, not run on hardware
//
#include <hip/hip_runtime.h>
#include <hip/hip_bf16.h>

typedef __bf16 bf16_t;
typedef __attribute__((ext_vector_type(16))) __bf16 v16bf;
typedef __attribute__((ext_vector_type(8)))  __bf16 v8bf;
typedef __attribute__((ext_vector_type(4)))  __bf16 v4bf;
typedef __attribute__((ext_vector_type(8)))  float  v8f;

#define N_SRC   50000
#define N_DST   50000
#define N_EDGES 800000
#define D_IN    256
#define NH      8
#define D_OUT   32
#define HD      256         // NH * D_OUT
#define REL_DIM 64
#define REL_OUT 512         // NH * 2 * D_OUT
#define LEAKY   0.01f

// ---------------- utility kernels ----------------

__global__ void k_fill_f32(float* __restrict__ p, float v, int n) {
    int i = blockIdx.x * blockDim.x + threadIdx.x;
    if (i < n) p[i] = v;
}

// 4 floats -> 4 bf16 per thread (n must be a multiple of 4; 12.8M is)
__global__ void k_f32_to_bf16_x4(const float* __restrict__ in, bf16_t* __restrict__ out, int n4) {
    int i = blockIdx.x * blockDim.x + threadIdx.x;
    if (i >= n4) return;
    float4 v = ((const float4*)in)[i];
    v4bf o = { (bf16_t)v.x, (bf16_t)v.y, (bf16_t)v.z, (bf16_t)v.w };
    ((v4bf*)out)[i] = o;
}

// Pack W[256 x 256] (f32, row-major KxN) into per-lane WMMA B fragments:
// Bp[((kt*16 + nt)*32 + lane)*16 + e] = (bf16) W[(kt*32 + (lane>>4)*16 + e) * 256 + nt*16 + (lane&15)]
// so each lane's 16-element fragment is one contiguous 32-byte chunk.
__global__ void k_pack_B(const float* __restrict__ W, bf16_t* __restrict__ Bp) {
    int i = blockIdx.x * blockDim.x + threadIdx.x;   // 65536 total
    if (i >= D_IN * HD) return;
    int e    = i & 15;
    int lane = (i >> 4) & 31;
    int nt   = (i >> 9) & 15;
    int kt   = i >> 13;                              // 0..7
    int col  = nt * 16 + (lane & 15);
    int k    = kt * 32 + (lane >> 4) * 16 + e;
    Bp[i] = (bf16_t)W[k * HD + col];
}

// rel_attn[j] = sum_k rel_emb[k] * W[k, j],  j < 512
__global__ void k_rel_attn(const float* __restrict__ rel_emb,
                           const float* __restrict__ W,
                           float* __restrict__ out) {
    int j = blockIdx.x * blockDim.x + threadIdx.x;
    if (j >= REL_OUT) return;
    float s = 0.f;
#pragma unroll 8
    for (int k = 0; k < REL_DIM; k++) s += rel_emb[k] * W[k * REL_OUT + j];
    out[j] = s;
}

// ---------------- WMMA GEMM: C[M x 256] = A[M x 256] * Bp(packed) ----------------
// block = 128 threads = 4 waves; block tile = 16(M) x 128(N); wave tile = 16 x 32
// (two 16x16 WMMA tiles sharing one A fragment). K loop fully unrolled: 16 WMMAs/wave.
__global__ __launch_bounds__(128)
void k_gemm_wmma(const bf16_t* __restrict__ A, const bf16_t* __restrict__ Bp,
                 float* __restrict__ C) {
    // row stride 264 bf16 = 528 B: 16-B aligned, rotates LDS banks by 4 DWORDs/row
    __shared__ bf16_t As[16][D_IN + 8];

    const int tid = threadIdx.x;
    const int m0  = blockIdx.x * 16;

    // stage the contiguous 16x256 (8 KB) A tile with 16-byte vector copies
    {
        const uint4* srcv = (const uint4*)(A + (size_t)m0 * D_IN);   // 512 chunks of 16 B
#pragma unroll
        for (int k = 0; k < 4; k++) {
            int j   = tid + k * 128;        // 0..511
            int row = j >> 5;
            int cc  = j & 31;
            *(uint4*)&As[row][cc * 8] = srcv[j];
        }
    }
    __syncthreads();

    const int wave = tid >> 5;
    const int lane = tid & 31;
    const int nt0  = blockIdx.y * 8 + wave * 2;      // first of two 16-col tiles
    const int rowA = lane & 15;
    const int kbA  = (lane >> 4) * 16;

    const v16bf* Bpv = (const v16bf*)Bp;             // fragment-major, 32 B each

    v8f acc0 = {};
    v8f acc1 = {};
#pragma unroll
    for (int kt = 0; kt < 8; kt++) {
        v8bf alo = *(const v8bf*)&As[rowA][kt * 32 + kbA];
        v8bf ahi = *(const v8bf*)&As[rowA][kt * 32 + kbA + 8];
        v16bf a = __builtin_shufflevector(alo, ahi,
                                          0, 1, 2, 3, 4, 5, 6, 7,
                                          8, 9, 10, 11, 12, 13, 14, 15);
        v16bf b0 = Bpv[(kt * 16 + nt0) * 32 + lane];
        v16bf b1 = Bpv[(kt * 16 + nt0 + 1) * 32 + lane];
        acc0 = __builtin_amdgcn_wmma_f32_16x16x32_bf16(false, a, false, b0,
                                                       (short)0, acc0, false, false);
        acc1 = __builtin_amdgcn_wmma_f32_16x16x32_bf16(false, a, false, b1,
                                                       (short)0, acc1, false, false);
    }

    // C/D layout: VGPR r -> M = r + 8*(lane>>4), N = lane&15
    const int cRow = m0 + (lane >> 4) * 8;
    const int cCol0 = nt0 * 16 + (lane & 15);
#pragma unroll
    for (int r = 0; r < 8; r++) {
        C[(size_t)(cRow + r) * HD + cCol0]      = acc0[r];
        C[(size_t)(cRow + r) * HD + cCol0 + 16] = acc1[r];
    }
}

// ---------------- attention scores ----------------
__global__ void k_attn_dot(const float* __restrict__ F,
                           const float* __restrict__ rel_attn, int rel_off,
                           float* __restrict__ e, int nNodes) {
    int i = blockIdx.x * blockDim.x + threadIdx.x;   // n*H + h
    if (i >= nNodes * NH) return;
    int n = i >> 3, h = i & 7;
    const float4* f = (const float4*)(F + (size_t)n * HD + h * D_OUT);
    const float4* r = (const float4*)(rel_attn + h * 2 * D_OUT + rel_off);
    float s = 0.f;
#pragma unroll
    for (int q = 0; q < 8; q++) {
        float4 a = f[q], b = r[q];
        s += a.x * b.x + a.y * b.y + a.z * b.z + a.w * b.w;
    }
    e[i] = s;
}

// float atomic max via signed-int max (v>=0) / unsigned min (v<0); target init to -inf
__device__ inline void atomicMaxF32(float* addr, float v) {
    if (v >= 0.f) atomicMax((int*)addr, __float_as_int(v));
    else          atomicMin((unsigned int*)addr, __float_as_uint(v));
}

__global__ void k_edge_logit_max(const float* __restrict__ e_src,
                                 const float* __restrict__ e_dst,
                                 const int* __restrict__ src_idx,
                                 const int* __restrict__ dst_idx,
                                 float* __restrict__ ebuf,
                                 float* __restrict__ segmax) {
    int i = blockIdx.x * blockDim.x + threadIdx.x;   // edge*H + h
    if (i >= N_EDGES * NH) return;
    int eidx = i >> 3, h = i & 7;
    int s = src_idx[eidx], d = dst_idx[eidx];
    float v = e_src[s * NH + h] + e_dst[d * NH + h];
    v = (v > 0.f) ? v : LEAKY * v;
    ebuf[i] = v;
    atomicMaxF32(&segmax[d * NH + h], v);
}

__global__ void k_fix_segmax(float* __restrict__ segmax, int n) {
    int i = blockIdx.x * blockDim.x + threadIdx.x;
    if (i >= n) return;
    float v = segmax[i];
    if (!(__builtin_fabsf(v) < __builtin_inff())) segmax[i] = 0.f;  // NaN/±inf -> 0
}

__global__ void k_exp_denom(const int* __restrict__ dst_idx,
                            const float* __restrict__ segmax,
                            float* __restrict__ ebuf,
                            float* __restrict__ denom) {
    int i = blockIdx.x * blockDim.x + threadIdx.x;
    if (i >= N_EDGES * NH) return;
    int eidx = i >> 3, h = i & 7;
    int d = dst_idx[eidx];
    float ex = __expf(ebuf[i] - segmax[d * NH + h]);
    ebuf[i] = ex;                          // reuse buffer: now holds exp values
    atomicAdd(&denom[d * NH + h], ex);
}

// msg scatter: agg[d, h, :] += fs[s, h, :] * a  (thread = one (edge, head), 32 floats)
__global__ void k_scatter(const float* __restrict__ fs,
                          const float* __restrict__ exbuf,
                          const float* __restrict__ denom,
                          const int* __restrict__ src_idx,
                          const int* __restrict__ dst_idx,
                          float* __restrict__ agg) {
    int i = blockIdx.x * blockDim.x + threadIdx.x;   // edge*H + h
    if (i >= N_EDGES * NH) return;
    int eidx = i >> 3, h = i & 7;
    int s = src_idx[eidx], d = dst_idx[eidx];
    float den = denom[d * NH + h];
    float a = exbuf[i] / ((den == 0.f) ? 1.f : den);
    const float4* fsv = (const float4*)(fs + (size_t)s * HD + h * D_OUT);
    __builtin_prefetch(fsv, 0, 1);                   // global_prefetch_b8
    float* aggp = agg + (size_t)d * HD + h * D_OUT;
#pragma unroll
    for (int q = 0; q < 8; q++) {
        float4 v = fsv[q];
        atomicAdd(aggp + q * 4 + 0, v.x * a);
        atomicAdd(aggp + q * 4 + 1, v.y * a);
        atomicAdd(aggp + q * 4 + 2, v.z * a);
        atomicAdd(aggp + q * 4 + 3, v.w * a);
    }
}

__global__ void k_relu_inplace(float* __restrict__ p, int n) {
    int i = blockIdx.x * blockDim.x + threadIdx.x;
    if (i >= n) return;
    float v = p[i];
    p[i] = (v > 0.f) ? v : 0.f;
}

// ---------------- launch ----------------

static inline size_t alignup(size_t x) { return (x + 255) & ~(size_t)255; }

extern "C" void kernel_launch(void* const* d_in, const int* in_sizes, int n_in,
                              void* d_out, int out_size, void* d_ws, size_t ws_size,
                              hipStream_t stream) {
    const float* feat_src = (const float*)d_in[0];
    const float* feat_dst = (const float*)d_in[1];
    const float* w_src    = (const float*)d_in[2];
    const float* w_dst    = (const float*)d_in[3];
    const float* rel_emb  = (const float*)d_in[4];
    const float* rel_w    = (const float*)d_in[5];
    const int*   src_idx  = (const int*)d_in[6];
    const int*   dst_idx  = (const int*)d_in[7];
    float* out = (float*)d_out;   // used as the aggregation buffer, relu'd in place

    // workspace carve-out
    char* ws = (char*)d_ws;
    size_t off = 0;
    float*  fs       = (float*)(ws + off); off = alignup(off + (size_t)N_SRC * HD * 4);
    float*  fd       = (float*)(ws + off); off = alignup(off + (size_t)N_DST * HD * 4);
    bf16_t* fsrc_bf  = (bf16_t*)(ws + off); off = alignup(off + (size_t)N_SRC * D_IN * 2);
    bf16_t* fdst_bf  = (bf16_t*)(ws + off); off = alignup(off + (size_t)N_DST * D_IN * 2);
    bf16_t* wsrc_bp  = (bf16_t*)(ws + off); off = alignup(off + (size_t)D_IN * HD * 2);
    bf16_t* wdst_bp  = (bf16_t*)(ws + off); off = alignup(off + (size_t)D_IN * HD * 2);
    float*  rel_attn = (float*)(ws + off); off = alignup(off + REL_OUT * 4);
    float*  e_src    = (float*)(ws + off); off = alignup(off + (size_t)N_SRC * NH * 4);
    float*  e_dst    = (float*)(ws + off); off = alignup(off + (size_t)N_DST * NH * 4);
    float*  segmax   = (float*)(ws + off); off = alignup(off + (size_t)N_DST * NH * 4);
    float*  denom    = (float*)(ws + off); off = alignup(off + (size_t)N_DST * NH * 4);
    float*  ebuf     = (float*)(ws + off); off = alignup(off + (size_t)N_EDGES * NH * 4);
    (void)ws_size; (void)n_in; (void)in_sizes; (void)out_size;

    const int nFeat = N_SRC * D_IN;          // 12,800,000
    const int nW    = D_IN * HD;             // 65,536
    const int nNH   = N_DST * NH;            // 400,000
    const int nEH   = N_EDGES * NH;          // 6,400,000

    // 1) precision conversion + B fragment packing for the WMMA GEMMs
    k_f32_to_bf16_x4<<<(nFeat / 4 + 255) / 256, 256, 0, stream>>>(feat_src, fsrc_bf, nFeat / 4);
    k_f32_to_bf16_x4<<<(nFeat / 4 + 255) / 256, 256, 0, stream>>>(feat_dst, fdst_bf, nFeat / 4);
    k_pack_B<<<(nW + 255) / 256, 256, 0, stream>>>(w_src, wsrc_bp);
    k_pack_B<<<(nW + 255) / 256, 256, 0, stream>>>(w_dst, wdst_bp);

    // 2) tiny relation-attention vector GEMV
    k_rel_attn<<<(REL_OUT + 255) / 256, 256, 0, stream>>>(rel_emb, rel_w, rel_attn);

    // 3) the two WMMA GEMMs: [50000 x 256] @ [256 x 256]
    dim3 gemm_grid(N_SRC / 16, HD / 128);    // (3125, 2)
    k_gemm_wmma<<<gemm_grid, 128, 0, stream>>>(fsrc_bf, wsrc_bp, fs);
    k_gemm_wmma<<<gemm_grid, 128, 0, stream>>>(fdst_bf, wdst_bp, fd);

    // 4) attention scores  (e_src uses rel_attn[:, D_OUT:], e_dst uses [:, :D_OUT])
    k_attn_dot<<<(nNH + 255) / 256, 256, 0, stream>>>(fs, rel_attn, D_OUT, e_src, N_SRC);
    k_attn_dot<<<(nNH + 255) / 256, 256, 0, stream>>>(fd, rel_attn, 0,     e_dst, N_DST);

    // 5) init accumulators (d_out/ws are poisoned, must be cleared every call)
    k_fill_f32<<<(nNH + 255) / 256, 256, 0, stream>>>(segmax, -__builtin_inff(), nNH);
    k_fill_f32<<<(nNH + 255) / 256, 256, 0, stream>>>(denom, 0.f, nNH);
    k_fill_f32<<<(nFeat + 255) / 256, 256, 0, stream>>>(out, 0.f, nFeat);

    // 6) edge softmax (segmented by dst)
    k_edge_logit_max<<<(nEH + 255) / 256, 256, 0, stream>>>(e_src, e_dst, src_idx, dst_idx,
                                                            ebuf, segmax);
    k_fix_segmax<<<(nNH + 255) / 256, 256, 0, stream>>>(segmax, nNH);
    k_exp_denom<<<(nEH + 255) / 256, 256, 0, stream>>>(dst_idx, segmax, ebuf, denom);

    // 7) message passing scatter + relu
    k_scatter<<<(nEH + 255) / 256, 256, 0, stream>>>(fs, ebuf, denom, src_idx, dst_idx, out);
    k_relu_inplace<<<(nFeat + 255) / 256, 256, 0, stream>>>(out, nFeat);
}
